// RewardLSTMModel_21234318311856
// MI455X (gfx1250) — matrix operand order repaired and checked by reading, not compiled, not run on hardware
//
#include <hip/hip_runtime.h>

#define INPUT_SIZE 65
#define HIDDEN     64
#define TSTEPS     256
#define BATCH      2048
#define BTILE      16
#define KPAD       160   // [ x: 0..64 (pad to 80) | h: 80..143 | bias@144 | pad ]
#define HOFF       80
#define BIASK      144   // constant-1.0 column in A, bias row in B
#define OUTW       7

typedef __attribute__((ext_vector_type(16))) __bf16 v16bf;
typedef __attribute__((ext_vector_type(8)))  __bf16 v8bf;
typedef __attribute__((ext_vector_type(8)))  float  v8f;

__device__ inline __bf16 f2bf(float f) {
    union { float f; unsigned u; } x; x.f = f;
    unsigned short s = (unsigned short)((x.u + 0x8000u) >> 16);  // round-to-nearest
    __bf16 b;
    __builtin_memcpy(&b, &s, sizeof(b));
    return b;
}
__device__ inline float bf2f(__bf16 b) {
    unsigned short s; __builtin_memcpy(&s, &b, sizeof(s));
    union { unsigned u; float f; } x; x.u = ((unsigned)s) << 16;
    return x.f;
}

// branch-free transcendentals: v_exp_f32 + v_rcp_f32, saturate correctly at +-inf
__device__ inline float sigmoidf_(float v) {
    return __builtin_amdgcn_rcpf(1.0f + __expf(-v));
}
__device__ inline float tanhf_(float v) {
    const float e = __expf(2.0f * v);               // inf -> rcp=0 -> +1 ; 0 -> -1
    return 1.0f - 2.0f * __builtin_amdgcn_rcpf(e + 1.0f);
}

__global__ __launch_bounds__(128, 1)
void lstm_fused_kernel(const float* __restrict__ x,
                       const float* __restrict__ W_ih,
                       const float* __restrict__ W_hh,
                       const float* __restrict__ b_ih,
                       const float* __restrict__ b_hh,
                       const float* __restrict__ fc_W,
                       const float* __restrict__ fc_b,
                       float* __restrict__ out)
{
    __shared__ alignas(16) __bf16 Abuf[2][BTILE * KPAD];  // double-buffered [x|h|1] tile

    const int tid    = threadIdx.x;
    const int wave   = tid >> 5;            // 0..3  (wave32)
    const int lane   = tid & 31;
    const int lane16 = lane & 15;
    const bool hiHalf = (lane & 16) != 0;
    const int b0 = blockIdx.x * BTILE;

    // ---- init both A buffers: zeros, except the constant-1 bias column ---
    // col 144 is never overwritten (x writes 0..64, h writes 80..143), so
    // the ks=4 WMMA adds bias via B's k==144 row for free every step.
    for (int i = tid; i < 2 * BTILE * KPAD; i += 128) {
        const int col = i % KPAD;
        ((__bf16*)Abuf)[i] = f2bf(col == BIASK ? 1.0f : 0.0f);
    }

    // ---- build register-resident B fragments:  B = [Wc | bias]^T ---------
    // wave w owns hidden columns [16w,16w+16) of each of the 4 gates (i,f,g,o)
    // B 32x16 bf16 layout: N = lane&15; lanes 0-15 carry K 0..7,16..23, lanes 16-31 K 8..15,24..31,
    // element e of v16bf is K = 32*ks + koffB + (e < 8 ? e : e + 8).
    v16bf bfrag[4][5];
    const int koffB = hiHalf ? 8 : 0;
#pragma unroll
    for (int g = 0; g < 4; ++g) {
        const int n = g * HIDDEN + wave * 16 + lane16;   // gate output column
#pragma unroll
        for (int ks = 0; ks < 5; ++ks) {
#pragma unroll
            for (int e = 0; e < 16; ++e) {
                const int k = ks * 32 + koffB + (e < 8 ? e : e + 8);       // combined-K index
                float v = 0.0f;
                if (k < INPUT_SIZE)                        v = W_ih[n * INPUT_SIZE + k];
                else if (k >= HOFF && k < HOFF + HIDDEN)   v = W_hh[n * HIDDEN + (k - HOFF)];
                else if (k == BIASK)                       v = b_ih[n] + b_hh[n];
                bfrag[g][ks][e] = f2bf(v);
            }
        }
    }

    // ---- persistent lane-local state (matches C/D 16x16 f32 layout) ------
    float c[8];
#pragma unroll
    for (int r = 0; r < 8; ++r) c[r] = 0.0f;

    const int arow  = lane16;               // A 16x32: M = lane&15
    const int koffA = hiHalf ? 8 : 0;       // hi-half lanes: K chunk +8
    const int hcol  = HOFF + wave * 16 + lane16;
    const int hrowb = hiHalf ? 8 : 0;       // D: lanes 16-31 hold M=8..15

    // ---- x-tile staging assignment: 8 threads per batch row --------------
    const int xrow = tid >> 3;              // 0..15
    const int xk   = (tid & 7) * 8;         // 0,8,...,56
    const float* xrowp = &x[((size_t)(b0 + xrow)) * TSTEPS * INPUT_SIZE];

    __syncthreads();

    for (int t = 0; t < TSTEPS; ++t) {
        __bf16* Ab = Abuf[t & 1];

        // -- stream x[:, t, :] tile into LDS as bf16 -----------------------
        // rows are only 4B-aligned in memory -> scalar b32 loads, but pack
        // the LDS side into a single 16B ds_store_b128 per thread.
        {
            const float* gp = xrowp + (size_t)t * INPUT_SIZE + xk;
            v8bf pk;
#pragma unroll
            for (int e = 0; e < 8; ++e) pk[e] = f2bf(gp[e]);
            *(v8bf*)&Ab[xrow * KPAD + xk] = pk;
            if ((tid & 7) == 0)
                Ab[xrow * KPAD + 64] = f2bf(gp[64 - xk]);
            if (t + 1 < TSTEPS)
                __builtin_prefetch(gp + INPUT_SIZE, 0, 1);   // next timestep
        }

        __syncthreads();

        // -- A fragments for this wave's 16 rows (2x ds_load_b128 each) ----
        v16bf afr[5];
#pragma unroll
        for (int ks = 0; ks < 5; ++ks) {
            const __bf16* p = &Ab[arow * KPAD + ks * 32 + koffA];
            v8bf lo = *(const v8bf*)p;          // K = base+0..7
            v8bf hi = *(const v8bf*)(p + 16);   // K = base+16..23
            afr[ks] = __builtin_shufflevector(lo, hi,
                        0,1,2,3,4,5,6,7,8,9,10,11,12,13,14,15);
        }

        // -- gates = [x_t | h_{t-1} | 1] x [Wc | bias]^T via bf16 WMMA -----
        // zero C init lets the backend use the inline-0 SRC2 form.
        v8f acc[4] = {};
#pragma unroll
        for (int ks = 0; ks < 5; ++ks) {
#pragma unroll
            for (int g = 0; g < 4; ++g) {
                acc[g] = __builtin_amdgcn_wmma_f32_16x16x32_bf16(
                            false, afr[ks], false, bfrag[g][ks],
                            (short)0, acc[g], false, false);
            }
        }

        // -- lane-local LSTM update; h_t -> next buffer (bf16) -------------
        __bf16* An = Abuf[(t + 1) & 1];
#pragma unroll
        for (int r = 0; r < 8; ++r) {
            const float iv = sigmoidf_(acc[0][r]);
            const float fv = sigmoidf_(acc[1][r]);
            const float gv = tanhf_(acc[2][r]);
            const float ov = sigmoidf_(acc[3][r]);
            c[r] = fv * c[r] + iv * gv;
            const float hv = ov * tanhf_(c[r]);
            An[(hrowb + r) * KPAD + hcol] = f2bf(hv);
        }
    }

    __syncthreads();

    // ---- head: out = sigmoid(h_T @ fc_W^T + fc_b) ------------------------
    // h_T sits as bf16 in Abuf[TSTEPS&1 == 0] columns HOFF..HOFF+63.
    const __bf16* Afin = Abuf[TSTEPS & 1];
    for (int i = tid; i < BTILE * OUTW; i += 128) {
        const int row = i / OUTW;
        const int oc  = i - row * OUTW;
        float a = fc_b[oc];
#pragma unroll
        for (int j = 0; j < HIDDEN; ++j)
            a += bf2f(Afin[row * KPAD + HOFF + j]) * fc_W[oc * HIDDEN + j];
        out[(size_t)(b0 + row) * OUTW + oc] = sigmoidf_(a);
    }
}

extern "C" void kernel_launch(void* const* d_in, const int* in_sizes, int n_in,
                              void* d_out, int out_size, void* d_ws, size_t ws_size,
                              hipStream_t stream) {
    const float* x    = (const float*)d_in[0];
    const float* W_ih = (const float*)d_in[1];
    const float* W_hh = (const float*)d_in[2];
    const float* b_ih = (const float*)d_in[3];
    const float* b_hh = (const float*)d_in[4];
    const float* fc_W = (const float*)d_in[5];
    const float* fc_b = (const float*)d_in[6];
    float* out = (float*)d_out;

    dim3 grid(BATCH / BTILE);   // 128 workgroups
    dim3 block(128);            // 4 wave32 per workgroup
    lstm_fused_kernel<<<grid, block, 0, stream>>>(x, W_ih, W_hh, b_ih, b_hh,
                                                  fc_W, fc_b, out);
}
